// STFT_8048768713403
// MI455X (gfx1250) — matrix operand order repaired and checked by reading, not compiled, not run on hardware
//
#include <hip/hip_runtime.h>
#include <hip/hip_bf16.h>

typedef __attribute__((ext_vector_type(16))) __bf16 v16bf;
typedef __attribute__((ext_vector_type(2)))  __bf16 v2bf;
typedef __attribute__((ext_vector_type(8)))  float  v8f;

#define N_FFT    1024
#define HOP      256
#define BATCH    16
#define SAMPLES  262144
#define N_FRAMES 1024
#define N_BINS   513
#define PAD_LEFT 384
#define NT16     36                      // 16-bin tiles covering 576 >= 513
#define KSTEPS   32                      // 1024 / 32
#define CHUNK_BF 2048                    // 32 lanes * 16 elems * 4 arrays (bf16)
#define WS_NEEDED ((size_t)KSTEPS * NT16 * CHUNK_BF * 2)   // bytes

// ---------------------------------------------------------------------------
// Kernel 1: precompute window-folded DFT fragments:
//   B'[k][n] = { w[k]*cos(2pi*k*n/N) , -w[k]*sin(2pi*k*n/N) }  (hi/lo bf16 split)
// Stored as B fragments in this code's contiguous k order (same as A): lanes 0-15 hold K=0-15 (col = lane&15),
// lanes 16-31 hold K=16-31; each lane's 16 bf16 contiguous (32 B).
// ---------------------------------------------------------------------------
__global__ __launch_bounds__(128) void build_b_kernel(
    const float* __restrict__ window, unsigned short* __restrict__ wsRaw) {
  __bf16* bw = reinterpret_cast<__bf16*>(wsRaw);
  const int kc = blockIdx.x;   // 0..31
  const int nt = blockIdx.y;   // 0..35
  __bf16* base = bw + ((size_t)(kc * NT16 + nt)) * CHUNK_BF;
  for (int p = threadIdx.x; p < 512; p += 128) {  // p = laneId*16 + e
    int laneId = p >> 4, e = p & 15;
    int k = kc * 32 + (laneId >> 4) * 16 + e;
    int n = nt * 16 + (laneId & 15);
    float w   = window[k];
    float ang = 6.28318530717958647692f * (float)((k * n) & (N_FFT - 1)) * (1.0f / (float)N_FFT);
    float c   = w * __cosf(ang);
    float ns  = -w * __sinf(ang);
    __bf16 ch = (__bf16)c;  __bf16 cl = (__bf16)(c  - (float)ch);
    __bf16 sh = (__bf16)ns; __bf16 sl = (__bf16)(ns - (float)sh);
    base[0 * 512 + p] = ch;
    base[1 * 512 + p] = cl;
    base[2 * 512 + p] = sh;
    base[3 * 512 + p] = sl;
  }
}

// ---------------------------------------------------------------------------
// Kernel 2: main GEMM. Grid (512, 9), block 128 = 4 wave32.
// WG owns 32 frames (shared A) x 64 bins; wave owns 16 bins, 2 M-subtiles.
// ---------------------------------------------------------------------------
__global__ __launch_bounds__(128) void stft_wmma_kernel(
    const float* __restrict__ x,
    const unsigned short* __restrict__ wsRaw,
    float* __restrict__ out)
{
  const __bf16* __restrict__ bw = reinterpret_cast<const __bf16*>(wsRaw);

  // A fragments, double-buffered: [buf][frag(M 0-15 / 16-31)][lane][8 u32 = 16 bf16]
  __shared__ __align__(32) unsigned aHi[2][2][32][8];
  __shared__ __align__(32) unsigned aLo[2][2][32][8];

  const int tid  = threadIdx.x;
  const int lane = tid & 31;
  const int wave = tid >> 5;

  const int mTile   = blockIdx.x;          // 0..511, 32 frames each
  const int b       = mTile >> 5;          // 32 M-tiles per batch
  const int mfBase  = (mTile & 31) << 5;   // frame base within batch
  const int nTile16 = blockIdx.y * 4 + wave;
  const int nBase   = nTile16 << 4;

  // Staging geometry: thread owns 8 consecutive K elements of one fragment row.
  const int sf      = tid >> 6;                 // fragment (M-subtile)
  const int sLane   = (tid >> 1) & 31;          // fragment lane
  const int e0b     = (tid & 1) << 3;           // element base: 0 or 8
  const int sHalf   = sLane >> 4;
  const int klbase  = sHalf * 16 + e0b;
  const int sFrame  = mfBase + (sLane & 15) + (sf << 4);
  const int sbase0  = sFrame * HOP + klbase - PAD_LEFT;  // + k0 each step; = 0 mod 8
  const int eSlot   = e0b >> 1;                 // u32 slot: 0 or 4

  // Only the first/last M-tile of each batch can touch the zero-pad region.
  const bool interior = (mfBase != 0) && (mfBase != (N_FRAMES - 32));

  v8f accR0 = {}, accI0 = {}, accR1 = {}, accI1 = {};
  const float* __restrict__ xb = x + (size_t)b * SAMPLES;

  for (int kc = 0; kc < KSTEPS; ++kc) {
    const int buf = kc & 1;
    const int sbase = sbase0 + kc * 32;

    // ---- stage 8 consecutive raw samples -> packed bf16 hi/lo ----
    float v[8];
    if (interior) {
      const float4* __restrict__ xa = (const float4*)(xb + sbase);
      float4 a0 = xa[0];
      float4 a1 = xa[1];
      v[0] = a0.x; v[1] = a0.y; v[2] = a0.z; v[3] = a0.w;
      v[4] = a1.x; v[5] = a1.y; v[6] = a1.z; v[7] = a1.w;
    } else {
      #pragma unroll
      for (int i = 0; i < 8; ++i) {
        int s  = sbase + i;
        int sc = s < 0 ? 0 : (s >= SAMPLES ? SAMPLES - 1 : s);
        float t = xb[sc];                 // unconditional clamped load
        v[i] = (s == sc) ? t : 0.0f;      // v_cndmask, no branch
      }
    }
    unsigned hu[4], lu[4];
    #pragma unroll
    for (int i = 0; i < 4; ++i) {
      float a = v[2 * i], c = v[2 * i + 1];
      __bf16 ha = (__bf16)a, hc = (__bf16)c;
      v2bf hp; hp[0] = ha; hp[1] = hc;
      v2bf lp; lp[0] = (__bf16)(a - (float)ha); lp[1] = (__bf16)(c - (float)hc);
      hu[i] = __builtin_bit_cast(unsigned, hp);
      lu[i] = __builtin_bit_cast(unsigned, lp);
    }
    *(uint4*)&aHi[buf][sf][sLane][eSlot] = make_uint4(hu[0], hu[1], hu[2], hu[3]);
    *(uint4*)&aLo[buf][sf][sLane][eSlot] = make_uint4(lu[0], lu[1], lu[2], lu[3]);

    // prefetch next K-step's B chunk toward the WGP (global_prefetch_b8)
    if (kc + 1 < KSTEPS)
      __builtin_prefetch((const void*)(bw + ((size_t)((kc + 1) * NT16 + nTile16)) * CHUNK_BF
                                       + (size_t)lane * 64), 0, 0);

    __syncthreads();  // single barrier per step (double-buffered staging)

    // ---- A fragments: 32 B contiguous per lane -> 2x ds_load_b128 each ----
    const v16bf ah0 = *(const v16bf*)&aHi[buf][0][lane][0];
    const v16bf al0 = *(const v16bf*)&aLo[buf][0][lane][0];
    const v16bf ah1 = *(const v16bf*)&aHi[buf][1][lane][0];
    const v16bf al1 = *(const v16bf*)&aLo[buf][1][lane][0];

    // ---- B fragments: precomputed, L2-resident, 32 B contiguous per lane ----
    const v16bf* __restrict__ bp =
        (const v16bf*)(bw + ((size_t)(kc * NT16 + nTile16)) * CHUNK_BF);
    const v16bf bch = bp[lane];
    const v16bf bcl = bp[32 + lane];
    const v16bf bsh = bp[64 + lane];
    const v16bf bsl = bp[96 + lane];

    // ---- 12 WMMAs: 3-term bf16 split x {real,imag} x {M-subtile 0,1} ----
    accR0 = __builtin_amdgcn_wmma_f32_16x16x32_bf16(false, ah0, false, bch, (short)0, accR0, false, false);
    accR0 = __builtin_amdgcn_wmma_f32_16x16x32_bf16(false, al0, false, bch, (short)0, accR0, false, false);
    accR0 = __builtin_amdgcn_wmma_f32_16x16x32_bf16(false, ah0, false, bcl, (short)0, accR0, false, false);
    accI0 = __builtin_amdgcn_wmma_f32_16x16x32_bf16(false, ah0, false, bsh, (short)0, accI0, false, false);
    accI0 = __builtin_amdgcn_wmma_f32_16x16x32_bf16(false, al0, false, bsh, (short)0, accI0, false, false);
    accI0 = __builtin_amdgcn_wmma_f32_16x16x32_bf16(false, ah0, false, bsl, (short)0, accI0, false, false);
    accR1 = __builtin_amdgcn_wmma_f32_16x16x32_bf16(false, ah1, false, bch, (short)0, accR1, false, false);
    accR1 = __builtin_amdgcn_wmma_f32_16x16x32_bf16(false, al1, false, bch, (short)0, accR1, false, false);
    accR1 = __builtin_amdgcn_wmma_f32_16x16x32_bf16(false, ah1, false, bcl, (short)0, accR1, false, false);
    accI1 = __builtin_amdgcn_wmma_f32_16x16x32_bf16(false, ah1, false, bsh, (short)0, accI1, false, false);
    accI1 = __builtin_amdgcn_wmma_f32_16x16x32_bf16(false, al1, false, bsh, (short)0, accI1, false, false);
    accI1 = __builtin_amdgcn_wmma_f32_16x16x32_bf16(false, ah1, false, bsl, (short)0, accI1, false, false);
  }

  // ---- store: rows r -> consecutive frames -> one 32 B vector store per acc ----
  const int n = nBase + (lane & 15);
  if (n < N_BINS) {
    const size_t imagOff = (size_t)BATCH * N_BINS * N_FRAMES;
    const int mf0 = mfBase + (lane >> 4) * 8;
    size_t o0 = ((size_t)b * N_BINS + n) * (size_t)N_FRAMES + (size_t)mf0;
    *(v8f*)&out[o0]                = accR0;   // 2x global_store_b128
    *(v8f*)&out[imagOff + o0]      = accI0;
    *(v8f*)&out[o0 + 16]           = accR1;
    *(v8f*)&out[imagOff + o0 + 16] = accI1;
  }
}

// ---------------------------------------------------------------------------
// Fallback (no workspace): compute B from LDS cos table in-loop.
// ---------------------------------------------------------------------------
__global__ __launch_bounds__(128) void stft_wmma_fallback(
    const float* __restrict__ x,
    const float* __restrict__ window,
    float* __restrict__ out)
{
  __shared__ float tabCos[N_FFT];
  __shared__ float tabWin[N_FFT];
  __shared__ float aTile[16][33];

  const int tid  = threadIdx.x;
  const int lane = tid & 31;
  const int wave = tid >> 5;

  const int mTile  = blockIdx.x;
  const int b      = mTile >> 6;
  const int mfBase = (mTile & 63) << 4;
  const int nBase  = blockIdx.y * 64 + wave * 16;

  const float twoPiOverN = 6.28318530717958647692f / (float)N_FFT;
  for (int i = tid; i < N_FFT; i += 128) {
    tabCos[i] = __cosf(twoPiOverN * (float)i);
    tabWin[i] = window[i];
  }
  __syncthreads();

  v8f accR = {}, accI = {};
  const int half = lane >> 4;
  const int mRow = lane & 15;
  const float* __restrict__ xb = x + (size_t)b * SAMPLES;
  const int nCol = nBase + mRow;

  for (int k0 = 0; k0 < N_FFT; k0 += 32) {
    #pragma unroll
    for (int e = 0; e < 4; ++e) {
      int idx = tid + e * 128;
      int row = idx >> 5, kk = idx & 31;
      int k = k0 + kk;
      int s = (mfBase + row) * HOP + k - PAD_LEFT;
      int sc = s < 0 ? 0 : (s >= SAMPLES ? SAMPLES - 1 : s);
      float t = xb[sc];
      aTile[row][kk] = ((s == sc) ? t : 0.0f) * tabWin[k];
    }
    __syncthreads();

    v16bf ah, al;
    #pragma unroll
    for (int e = 0; e < 16; ++e) {
      int kl = half * 16 + e;
      float a = aTile[mRow][kl];
      __bf16 h = (__bf16)a;
      ah[e] = h;
      al[e] = (__bf16)(a - (float)h);
    }
    v16bf bch, bcl, bsh, bsl;
    #pragma unroll
    for (int e = 0; e < 16; ++e) {
      int k  = k0 + half * 16 + e;
      int ai = (k * nCol) & (N_FFT - 1);
      float c  = tabCos[ai];
      float ns = -tabCos[(ai + 768) & (N_FFT - 1)];
      __bf16 ch = (__bf16)c;  bch[e] = ch; bcl[e] = (__bf16)(c  - (float)ch);
      __bf16 sh = (__bf16)ns; bsh[e] = sh; bsl[e] = (__bf16)(ns - (float)sh);
    }
    accR = __builtin_amdgcn_wmma_f32_16x16x32_bf16(false, ah, false, bch, (short)0, accR, false, false);
    accR = __builtin_amdgcn_wmma_f32_16x16x32_bf16(false, al, false, bch, (short)0, accR, false, false);
    accR = __builtin_amdgcn_wmma_f32_16x16x32_bf16(false, ah, false, bcl, (short)0, accR, false, false);
    accI = __builtin_amdgcn_wmma_f32_16x16x32_bf16(false, ah, false, bsh, (short)0, accI, false, false);
    accI = __builtin_amdgcn_wmma_f32_16x16x32_bf16(false, al, false, bsh, (short)0, accI, false, false);
    accI = __builtin_amdgcn_wmma_f32_16x16x32_bf16(false, ah, false, bsl, (short)0, accI, false, false);
    __syncthreads();
  }

  const int n = nBase + (lane & 15);
  if (n < N_BINS) {
    const size_t imagOff = (size_t)BATCH * N_BINS * N_FRAMES;
    const int mf0 = mfBase + half * 8;
    #pragma unroll
    for (int r = 0; r < 8; ++r) {
      size_t o = ((size_t)b * N_BINS + n) * (size_t)N_FRAMES + (size_t)(mf0 + r);
      out[o]           = accR[r];
      out[imagOff + o] = accI[r];
    }
  }
}

extern "C" void kernel_launch(void* const* d_in, const int* in_sizes, int n_in,
                              void* d_out, int out_size, void* d_ws, size_t ws_size,
                              hipStream_t stream) {
  (void)in_sizes; (void)n_in; (void)out_size;
  const float* x      = (const float*)d_in[0];
  const float* window = (const float*)d_in[1];
  float* out          = (float*)d_out;

  if (ws_size >= WS_NEEDED && d_ws != nullptr) {
    unsigned short* ws = (unsigned short*)d_ws;
    build_b_kernel<<<dim3(KSTEPS, NT16, 1), dim3(128, 1, 1), 0, stream>>>(window, ws);
    stft_wmma_kernel<<<dim3(512, 9, 1), dim3(128, 1, 1), 0, stream>>>(x, ws, out);
  } else {
    stft_wmma_fallback<<<dim3(1024, 9, 1), dim3(128, 1, 1), 0, stream>>>(x, window, out);
  }
}
